// MuyGP_824633721002
// MI455X (gfx1250) — matrix operand order repaired and checked by reading, not compiled, not run on hardware
//
#include <hip/hip_runtime.h>

// Problem constants (from reference setup_inputs)
#define NB      1024      // batch
#define NTRAIN  100000    // training points
#define DIM     256       // feature dim
#define KNN     128       // neighbors
#define NSPLIT  16        // N-dimension slices for fused top-k
#define SLICE   6250      // NTRAIN / NSPLIT
#define NCHUNK  49        // ceil(SLICE / 128)
#define LDA     132       // padded leading dim for 128x128 LDS matrix
#define TXLD    264       // padded leading dim (bf16) for LDS tiles (512B row + 16B pad)
#define FINF    (__builtin_inff())

typedef __attribute__((ext_vector_type(16))) __bf16        v16bf;
typedef __attribute__((ext_vector_type(8)))  float         v8f;
typedef __attribute__((ext_vector_type(4)))  unsigned int  u32x4;
typedef __attribute__((ext_vector_type(8)))  int           i32x8;
typedef __attribute__((ext_vector_type(4)))  int           i32x4;

// ---------------------------------------------------------------------------
// Tensor Data Mover descriptor helpers (CDNA5 D#, CDNA5 ISA)
// ---------------------------------------------------------------------------
__device__ __forceinline__ unsigned lds_byte_off(const void* p) {
  // Flat-aperture rule: for LDS-space addresses, the LDS byte offset is the low 32 bits of the address
  return (unsigned)(size_t)p;
}

__device__ __forceinline__ u32x4 tdm_g0(unsigned ctrl, unsigned ldsaddr, const void* gaddr) {
  unsigned long long ga = (unsigned long long)(size_t)gaddr;
  u32x4 g0;
  g0[0] = ctrl;                                              // count=1 (+ gather bits)
  g0[1] = ldsaddr;                                           // lds_addr (bytes)
  g0[2] = (unsigned)ga;                                      // global_addr[31:0]
  g0[3] = (unsigned)((ga >> 32) & 0x01ffffffu) | (2u << 30); // global_addr[56:32], type=2
  return g0;
}

// group1 for: 8-byte elements, row = 64 elems (512B), LDS pad 4 dwords per 128 dwords
__device__ __forceinline__ i32x8 tdm_g1(unsigned tdim1, unsigned tile1) {
  const unsigned td0 = 64, tile0 = 64, s0 = 64;
  i32x8 g1;
  g1[0] = (int)((3u << 16) | (1u << 20) | (6u << 22) | (3u << 25));
  g1[1] = (int)(td0 << 16);                                   // tensor_dim0[15:0]
  g1[2] = (int)((td0 >> 16) | ((tdim1 & 0xffffu) << 16));     // tdim0 hi | tdim1 lo
  g1[3] = (int)(((tdim1 >> 16) & 0xffffu) | (tile0 << 16));   // tdim1 hi | tile_dim0
  g1[4] = (int)(tile1 & 0xffffu);                             // tile_dim1 | tile_dim2=0
  g1[5] = (int)s0;                                            // tensor_dim0_stride lo
  g1[6] = 0;                                                  // stride0 hi | stride1 lo
  g1[7] = 0;
  return g1;
}

#if __has_include(<hip/amd_detail/amd_gfx1250_TDM.h>)
// amdgpu-toolchain (clang-23 / therock-10.0 headers): 6-arg builtin
__device__ __forceinline__ void tdm_issue(u32x4 g0, i32x8 g1, i32x4 g2, i32x4 g3) {
  i32x8 g4 = {0, 0, 0, 0, 0, 0, 0, 0};
  __builtin_amdgcn_tensor_load_to_lds(g0, g1, g2, g3, g4, 0);
}
#else
// ROCm 7.2 (clang-22): 5-arg builtin
__device__ __forceinline__ void tdm_issue(u32x4 g0, i32x8 g1, i32x4 g2, i32x4 g3) {
  __builtin_amdgcn_tensor_load_to_lds(g0, g1, g2, g3, 0);
}
#endif

// 2D contiguous tile: rows x 512B from gsrc -> LDS (row-padded to 528B)
__device__ __forceinline__ void tdm_load_chunk(const __bf16* gsrc, unsigned ldsaddr, int rows) {
  u32x4 g0 = tdm_g0(1u, ldsaddr, gsrc);
  i32x8 g1 = tdm_g1((unsigned)rows, 128u);
  i32x4 gz = {0, 0, 0, 0};
  tdm_issue(g0, g1, gz, gz);
}

// gather-mode tile: 8 rows (32-bit indices) x 512B from tensor base -> LDS
__device__ __forceinline__ void tdm_gather8(const __bf16* gbase, unsigned ldsaddr,
                                            const int* idx8) {
  u32x4 g0 = tdm_g0(1u | (1u << 30) | (1u << 31), ldsaddr, gbase);  // gather, 32b idx
  i32x8 g1 = tdm_g1((unsigned)NTRAIN, 8u);                           // tile_dim1 = #indices
  i32x4 g2, g3;
#pragma unroll
  for (int u = 0; u < 4; ++u) { g2[u] = idx8[u]; g3[u] = idx8[4 + u]; }
  tdm_issue(g0, g1, g2, g3);
}

__device__ __forceinline__ float wredsum(float v) {
#pragma unroll
  for (int o = 16; o; o >>= 1) v += __shfl_xor(v, o, 32);
  return v;
}

// gfx1250 WMMA bf16 A-fragment (16x32), per ISA layout:
// lane L holds row m=L&15; K = kb*32 + 8*(L>>4) + e + (e>=8 ? 8 : 0)
__device__ __forceinline__ v16bf load_afrag(const __bf16* base, int ld, int kb, int lane) {
  v16bf a;
  const __bf16* p = base + (lane & 15) * ld + kb * 32 + ((lane >> 4) << 3);
#pragma unroll
  for (int e = 0; e < 8; ++e) a[e] = p[e];
#pragma unroll
  for (int e = 8; e < 16; ++e) a[e] = p[e + 8];
  return a;
}

// gfx1250 WMMA bf16 B-fragment (32x16): lane L holds col n=L&15,
// K = kb*32 + 8*(L>>4) + e + (e>=8 ? 8 : 0)
__device__ __forceinline__ v16bf load_bfrag(const __bf16* base, int ld, int kb, int lane) {
  v16bf b;
  const __bf16* p = base + (lane & 15) * ld + kb * 32 + ((lane >> 4) << 3);
#pragma unroll
  for (int e = 0; e < 8; ++e) b[e] = p[e];
#pragma unroll
  for (int e = 8; e < 16; ++e) b[e] = p[e + 8];
  return b;
}

// ---------------------------------------------------------------------------
// K1: trainX fp32 -> bf16, plus squared row norms. One wave per row.
// ---------------------------------------------------------------------------
__global__ __launch_bounds__(256) void k1_prep(const float* __restrict__ tX,
                                               __bf16* __restrict__ txbf,
                                               float* __restrict__ tnorm) {
  int row  = blockIdx.x * 8 + (threadIdx.x >> 5);
  int lane = threadIdx.x & 31;
  const float4* src = (const float4*)(tX + (size_t)row * DIM);
  float s = 0.f;
#pragma unroll
  for (int v = 0; v < 2; ++v) {
    float4 f = src[lane * 2 + v];
    __bf16* dst = txbf + (size_t)row * DIM + lane * 8 + v * 4;
    dst[0] = (__bf16)f.x; dst[1] = (__bf16)f.y;
    dst[2] = (__bf16)f.z; dst[3] = (__bf16)f.w;
    s += f.x * f.x + f.y * f.y + f.z * f.z + f.w * f.w;
  }
  s = wredsum(s);
  if (lane == 0) tnorm[row] = s;
}

// ---------------------------------------------------------------------------
// Wave-cooperative reservoir top-k update for one query row.
// ---------------------------------------------------------------------------
__device__ __forceinline__ void topk_row(float* resDr, int* resIr, const float* chDr,
                                         int base, float& thr, int& mpos, int lane) {
#pragma unroll 1
  for (int pass = 0; pass < 4; ++pass) {
    int   c = pass * 32 + lane;
    float d = chDr[c];
    unsigned long long mk = __ballot(d < thr);
    while (mk) {
      int src = __ffsll((unsigned long long)mk) - 1;
      mk &= mk - 1;
      float dc = __shfl(d, src, 32);
      if (dc < thr) {                 // re-check: thr may have tightened
        if (lane == 0) { resDr[mpos] = dc; resIr[mpos] = base + pass * 32 + src; }
        float mv = -1.f; int mp = 0;
#pragma unroll
        for (int j = 0; j < 4; ++j) {
          int jj = j * 32 + lane;
          float v2 = resDr[jj];
          if (v2 > mv) { mv = v2; mp = jj; }
        }
#pragma unroll
        for (int o = 16; o; o >>= 1) {
          float ov = __shfl_xor(mv, o, 32);
          int   op = __shfl_xor(mp, o, 32);
          if (ov > mv) { mv = ov; mp = op; }
        }
        thr = mv; mpos = mp;
      }
    }
  }
}

// ---------------------------------------------------------------------------
// K2: fused bf16-WMMA distance GEMM + per-slice top-128, with double-buffered
// TDM (tensor_load_to_lds) streaming of trainX chunks.
// Grid: (NB/16, NSPLIT). Block: 256 (8 waves).
// ---------------------------------------------------------------------------
__global__ __launch_bounds__(256) void k2_topk(const float* __restrict__ x,
                                               const __bf16* __restrict__ txbf,
                                               const float* __restrict__ tnorm,
                                               float* __restrict__ candD,
                                               int* __restrict__ candI) {
  __shared__ __bf16 xt[16][TXLD];
  __shared__ __bf16 tx[2][128][TXLD];
  __shared__ float  chD[16][128];
  __shared__ float  resD[16][128];
  __shared__ int    resI[16][128];
  __shared__ float  xnS[16];
  __shared__ float  tnS[128];

  const int t = threadIdx.x;
  const int lane = t & 31;
  const int w = t >> 5;
  const int mtile = blockIdx.x;
  const int slice = blockIdx.y;

  if (t < 16) xnS[t] = 0.f;
  for (int e = t; e < 16 * 128; e += 256) { (&resD[0][0])[e] = FINF; (&resI[0][0])[e] = 0; }
  __syncthreads();

  // stage 16 query rows: fp32 -> bf16 in LDS + fp32 norms
  {
    int r = t >> 4, seg = t & 15;
    const float* xr = x + (size_t)(mtile * 16 + r) * DIM + seg * 16;
    float s = 0.f;
#pragma unroll
    for (int u = 0; u < 16; ++u) { float f = xr[u]; xt[r][seg * 16 + u] = (__bf16)f; s += f * f; }
    atomicAdd(&xnS[r], s);
  }

  // TDM preload of chunk 0 into buffer 0 (wave 0 issues; TENSORcnt is per-wave)
  if (w == 0) {
    tdm_load_chunk(txbf + (size_t)(slice * SLICE) * DIM,
                   lds_byte_off(&tx[0][0][0]), 128);
  }

  float thr0 = FINF, thr1 = FINF;
  int   mp0 = 0,    mp1 = 0;

  for (int chunk = 0; chunk < NCHUNK; ++chunk) {
    const int buf   = chunk & 1;
    const int base  = slice * SLICE + chunk * 128;
    const int valid = (SLICE - chunk * 128 < 128) ? (SLICE - chunk * 128) : 128;

    if (w == 0) {
      if (chunk + 1 < NCHUNK) {   // prefetch next chunk via TDM into other buffer
        int nbase = slice * SLICE + (chunk + 1) * 128;
        int nrows = (SLICE - (chunk + 1) * 128 < 128) ? (SLICE - (chunk + 1) * 128) : 128;
        tdm_load_chunk(txbf + (size_t)nbase * DIM,
                       lds_byte_off(&tx[buf ^ 1][0][0]), nrows);
        __builtin_amdgcn_s_wait_tensorcnt((short)1);  // in-order: current chunk landed
      } else {
        __builtin_amdgcn_s_wait_tensorcnt((short)0);
      }
    }
    if (t < 128) tnS[t] = (t < valid) ? tnorm[base + t] : FINF;
    __syncthreads();

    v8f acc = {0.f, 0.f, 0.f, 0.f, 0.f, 0.f, 0.f, 0.f};
#pragma unroll
    for (int kb = 0; kb < 8; ++kb) {
      v16bf af = load_afrag(&xt[0][0], TXLD, kb, lane);
      v16bf bf = load_bfrag(&tx[buf][w * 16][0], TXLD, kb, lane);
      acc = __builtin_amdgcn_wmma_f32_16x16x32_bf16(false, af, false, bf,
                                                    (short)0, acc, false, false);
    }
    {
      int ml = (lane >> 4) << 3, nl = lane & 15;
      int n = w * 16 + nl;
      float tn = tnS[n];
#pragma unroll
      for (int v = 0; v < 8; ++v) {
        int m = ml + v;
        float sq = xnS[m] + tn - 2.f * acc[v];
        float dd = (n < valid) ? (sq > 0.f ? sqrtf(sq) : 0.f) : FINF;
        chD[m][n] = dd;
      }
    }
    __syncthreads();
    topk_row(resD[2 * w],     resI[2 * w],     chD[2 * w],     base, thr0, mp0, lane);
    topk_row(resD[2 * w + 1], resI[2 * w + 1], chD[2 * w + 1], base, thr1, mp1, lane);
    __syncthreads();
  }

  for (int e = t; e < 2048; e += 256) {
    int r = e >> 7, j = e & 127;
    size_t m = (size_t)mtile * 16 + r;
    candD[(m * NSPLIT + slice) * KNN + j] = resD[r][j];
    candI[(m * NSPLIT + slice) * KNN + j] = resI[r][j];
  }
}

// ---------------------------------------------------------------------------
// K3: merge NSPLIT*128 = 2048 candidates per query row -> 128 smallest
// (bitonic sort in LDS). One block per query row.
// ---------------------------------------------------------------------------
__global__ __launch_bounds__(256) void k3_merge(const float* __restrict__ candD,
                                                const int* __restrict__ candI,
                                                int* __restrict__ nbIdx,
                                                float* __restrict__ nbDist) {
  __shared__ float sd[2048];
  __shared__ int   si[2048];
  const int b = blockIdx.x, t = threadIdx.x;
  for (int e = t; e < 2048; e += 256) {
    sd[e] = candD[(size_t)b * 2048 + e];
    si[e] = candI[(size_t)b * 2048 + e];
  }
  __syncthreads();
  for (int k = 2; k <= 2048; k <<= 1)
    for (int j = k >> 1; j > 0; j >>= 1) {
      for (int e = t; e < 2048; e += 256) {
        int ixj = e ^ j;
        if (ixj > e) {
          bool up = ((e & k) == 0);
          float a = sd[e], c = sd[ixj];
          if ((a > c) == up) {
            sd[e] = c; sd[ixj] = a;
            int ti = si[e]; si[e] = si[ixj]; si[ixj] = ti;
          }
        }
      }
      __syncthreads();
    }
  if (t < 128) { nbIdx[b * KNN + t] = si[t]; nbDist[b * KNN + t] = sd[t]; }
}

// ---------------------------------------------------------------------------
// K4: TDM gather-mode neighbor fetch -> WMMA Gram (128x128) -> kernel
// transform -> LDS Cholesky -> triangular solves -> y, yVar. 1 block / batch.
// ---------------------------------------------------------------------------
__global__ __launch_bounds__(256) void k4_solve(const __bf16* __restrict__ txbf,
                                                const float* __restrict__ trainy,
                                                const float* __restrict__ noise,
                                                const int* __restrict__ nbIdx,
                                                const float* __restrict__ nbDist,
                                                const float* __restrict__ lP,
                                                const float* __restrict__ aP,
                                                const float* __restrict__ ymP,
                                                float* __restrict__ out) {
  __shared__ __bf16 nXl[128][TXLD];
  __shared__ float  A[128 * LDA];
  __shared__ float  ns[128], cS[128], nyS[128], zS[128], wS[128];
  __shared__ int    idxS[128];

  const int b = blockIdx.x, t = threadIdx.x;
  const int lane = t & 31, w = t >> 5;
  const float lv = expf(lP[0]);
  const float av = expf(aP[0]);
  const float ym = ymP[0];
  const float isc = 1.0f / (16.0f * lv);   // sqrt(DIM)=16; x/lv scaling folded in

  if (t < 128) idxS[t] = nbIdx[b * KNN + t];
  __syncthreads();

  // gather 128 scattered neighbor rows via TDM gather mode: 16 x (8 rows)
  if (w == 0) {
#pragma unroll 1
    for (int g = 0; g < 16; ++g) {
      int idx8[8];
#pragma unroll
      for (int u = 0; u < 8; ++u) idx8[u] = idxS[g * 8 + u];
      tdm_gather8(txbf, lds_byte_off(&nXl[g * 8][0]), idx8);
    }
    __builtin_amdgcn_s_wait_tensorcnt((short)0);
  }
  if (t < 128) {
    cS[t]  = av * expf(-nbDist[b * KNN + t] * isc);                 // crossCov
    nyS[t] = trainy[idxS[t]] + 0.01f * noise[b * KNN + t] - ym;     // ny
  }
  __syncthreads();

  // Gram G = nX * nX^T via WMMA; wave w owns row-tile w (8 N-tiles x 8 K-steps)
  for (int nt = 0; nt < 8; ++nt) {
    v8f acc = {0.f, 0.f, 0.f, 0.f, 0.f, 0.f, 0.f, 0.f};
#pragma unroll
    for (int kb = 0; kb < 8; ++kb) {
      v16bf af = load_afrag(&nXl[w * 16][0], TXLD, kb, lane);
      v16bf bf = load_bfrag(&nXl[nt * 16][0], TXLD, kb, lane);
      acc = __builtin_amdgcn_wmma_f32_16x16x32_bf16(false, af, false, bf,
                                                    (short)0, acc, false, false);
    }
    int ml = (lane >> 4) << 3, nl = lane & 15;
#pragma unroll
    for (int v = 0; v < 8; ++v)
      A[(w * 16 + ml + v) * LDA + nt * 16 + nl] = acc[v];
  }
  __syncthreads();
  if (t < 128) ns[t] = A[t * LDA + t];
  __syncthreads();

  // auto_ij = av * exp(-sqrt(max(ni+nj-2G,0)) / (16*lv)); diagonal -> av exactly
  for (int e = t; e < 128 * 128; e += 256) {
    int i = e >> 7, j = e & 127;
    float sq = ns[i] + ns[j] - 2.f * A[i * LDA + j];
    float d = sq > 0.f ? sqrtf(sq) : 0.f;
    A[i * LDA + j] = av * expf(-d * isc);
  }
  __syncthreads();

  // in-place Cholesky (lower) of A
  for (int k = 0; k < 128; ++k) {
    float diag = A[k * LDA + k];
    float inv = rsqrtf(fmaxf(diag, 1e-30f));
    __syncthreads();
    if (t == 0) A[k * LDA + k] = diag * inv;         // sqrt(diag)
    int i = k + 1 + t;
    float lik = 0.f;
    if (i < 128) { lik = A[i * LDA + k] * inv; A[i * LDA + k] = lik; }
    __syncthreads();
    if (i < 128)
      for (int j = k + 1; j <= i; ++j) A[i * LDA + j] -= lik * A[j * LDA + k];
    __syncthreads();
  }

  // wave 0: forward/backward triangular solves + output dots
  if (t < 32) {
    for (int k = 0; k < 128; ++k) {      // L z = c
      float p = 0.f;
      for (int j = t; j < k; j += 32) p += A[k * LDA + j] * zS[j];
      p = wredsum(p);
      if (t == 0) zS[k] = (cS[k] - p) / A[k * LDA + k];
    }
    for (int k = 127; k >= 0; --k) {     // L^T w = z
      float p = 0.f;
      for (int i = k + 1 + t; i < 128; i += 32) p += A[i * LDA + k] * wS[i];
      p = wredsum(p);
      if (t == 0) wS[k] = (zS[k] - p) / A[k * LDA + k];
    }
    float yp = 0.f, vp = 0.f;
    for (int j = t; j < 128; j += 32) { yp += wS[j] * nyS[j]; vp += wS[j] * cS[j]; }
    yp = wredsum(yp); vp = wredsum(vp);
    if (t == 0) { out[b] = yp + ym; out[NB + b] = av - vp; }
  }
}

// ---------------------------------------------------------------------------
extern "C" void kernel_launch(void* const* d_in, const int* in_sizes, int n_in,
                              void* d_out, int out_size, void* d_ws, size_t ws_size,
                              hipStream_t stream) {
  const float* x      = (const float*)d_in[0];
  const float* trainX = (const float*)d_in[1];
  const float* trainy = (const float*)d_in[2];
  const float* noise  = (const float*)d_in[3];
  const float* lP     = (const float*)d_in[4];
  const float* aP     = (const float*)d_in[5];
  const float* ymP    = (const float*)d_in[6];
  float* out = (float*)d_out;
  char* ws = (char*)d_ws;
  (void)in_sizes; (void)n_in; (void)out_size; (void)ws_size;

  size_t off = 0;
  auto alloc = [&](size_t bytes) {
    size_t o = off;
    off = (off + bytes + 255) & ~(size_t)255;
    return o;
  };
  __bf16* txbf  = (__bf16*)(ws + alloc((size_t)NTRAIN * DIM * 2));
  float*  tnorm = (float*)(ws + alloc((size_t)NTRAIN * sizeof(float)));
  float*  candD = (float*)(ws + alloc((size_t)NB * NSPLIT * KNN * sizeof(float)));
  int*    candI = (int*)(ws + alloc((size_t)NB * NSPLIT * KNN * sizeof(int)));
  int*    nbIdx = (int*)(ws + alloc((size_t)NB * KNN * sizeof(int)));
  float*  nbD   = (float*)(ws + alloc((size_t)NB * KNN * sizeof(float)));

  k1_prep<<<NTRAIN / 8, 256, 0, stream>>>(trainX, txbf, tnorm);
  k2_topk<<<dim3(NB / 16, NSPLIT), 256, 0, stream>>>(x, txbf, tnorm, candD, candI);
  k3_merge<<<NB, 256, 0, stream>>>(candD, candI, nbIdx, nbD);
  k4_solve<<<NB, 256, 0, stream>>>(txbf, trainy, noise, nbIdx, nbD, lP, aP, ymP, out);
}
